// UNet_fuselang_36524401885206
// MI455X (gfx1250) — matrix operand order repaired and checked by reading, not compiled, not run on hardware
//
#include <hip/hip_runtime.h>
#include <hip/hip_bf16.h>
#include <stdint.h>

typedef __attribute__((ext_vector_type(16))) _Float16 v16h;
typedef __attribute__((ext_vector_type(8)))  _Float16 v8h;
typedef __attribute__((ext_vector_type(8)))  float    v8f;

union AFrag { v16h v; v8h h[2]; };

// ---------------------------------------------------------------------------
// Generic implicit-GEMM conv / deconv.
// mode 0: 3x3x3 conv, stride 1, pad 1 (SAME)      src = o + t - 1
// mode 1: 1x1x1 conv                               src = o
// mode 2: 3x3x3 transposed conv, stride 2 (SAME)   src = (o+t-2)/2 if even
// Channels: C1p from in1 ++ C2p from in2 (both padded to multiples of 32).
// ---------------------------------------------------------------------------
struct ConvParams {
  const _Float16* in1;
  const _Float16* in2;
  const _Float16* wp;    // packed B fragments: [NT][KT][32 lanes][16 halfs]
  const _Float16* zero;  // zero page for out-of-bounds taps
  _Float16* out;         // [M, Cout] fp16
  int inD, inH, inW;
  int outD, outH, outW;
  int lw, lh, ld;        // log2 of out dims
  int C1p, C2p, Cinp, Cout;
  int taps, mode, bcast, relu;
  int KT;                // taps * Cinp / 32
};

template <int NB>
__global__ __launch_bounds__(256)
void wmma_conv(ConvParams p) {
  const int lane   = threadIdx.x & 31;
  const int wave   = threadIdx.x >> 5;
  const int mtile  = blockIdx.x * 8 + wave;      // 8 waves / block
  const int ntBase = blockIdx.y * NB;
  const int mBase  = mtile << 4;
  const int grp    = lane >> 4;
  const int ml     = lane & 15;

  // This lane's A-row (output position): A layout M = lane & 15.
  const int m  = mBase + ml;
  const int ow = m & (p.outW - 1);
  const int oh = (m >> p.lw) & (p.outH - 1);
  const int od = (m >> (p.lw + p.lh)) & (p.outD - 1);
  const int b  = m >> (p.lw + p.lh + p.ld);

  v8f acc[NB];
#pragma unroll
  for (int j = 0; j < NB; ++j) acc[j] = (v8f){};

  // Per-N-tile B streams, contiguous in kt (1024 B per chunk step).
  const _Float16* wpj[NB];
#pragma unroll
  for (int j = 0; j < NB; ++j)
    wpj[j] = p.wp + ((size_t)(ntBase + j) * p.KT) * 512 + lane * 16;

  const int aoff = grp << 3;   // halfs 0-7 -> K grp*8.., halfs 8-15 -> +16

  int td = 0, th = 0, tw = 0;  // wave-uniform incremental tap coordinates
  for (int tap = 0; tap < p.taps; ++tap) {
    int sd, sh, sw;
    bool ok = true;
    if (p.mode == 2) {
      const int dd = od + td - 2, hh = oh + th - 2, ww = ow + tw - 2;
      const int orall = dd | hh | ww;
      ok = (orall >= 0) && ((orall & 1) == 0);
      sd = dd >> 1; sh = hh >> 1; sw = ww >> 1;
      ok = ok && (sd < p.inD) && (sh < p.inH) && (sw < p.inW);
    } else if (p.mode == 0) {
      sd = od + td - 1; sh = oh + th - 1; sw = ow + tw - 1;
      ok = ((unsigned)sd < (unsigned)p.inD) &
           ((unsigned)sh < (unsigned)p.inH) &
           ((unsigned)sw < (unsigned)p.inW);
    } else { sd = od; sh = oh; sw = ow; }
    // advance (tw,th,td) with carries (no divides)
    if (++tw == 3) { tw = 0; if (++th == 3) { th = 0; ++td; } }

    const size_t vox = ok ? ((((size_t)b * p.inD + sd) * p.inH + sh) * p.inW + sw)
                          : (size_t)0;

    int kt = tap * (p.Cinp >> 5);

    // prefetch the (linear) B streams ahead
#pragma unroll
    for (int j = 0; j < NB; ++j)
      __builtin_prefetch(wpj[j] + (size_t)(kt + 16) * 512, 0, 3);

    // segment 1: C1p channels from in1
    {
      const _Float16* base = ok ? (p.in1 + vox * p.C1p + aoff) : (p.zero + aoff);
      for (int c = 0; c < p.C1p; c += 32) {
        AFrag a;
        a.h[0] = *(const v8h*)(base + c);
        a.h[1] = *(const v8h*)(base + c + 16);
#pragma unroll
        for (int j = 0; j < NB; ++j) {
          v16h bf = *(const v16h*)(wpj[j] + (size_t)kt * 512);
          acc[j] = __builtin_amdgcn_wmma_f32_16x16x32_f16(
              false, a.v, false, bf, (short)0, acc[j], false, false);
        }
        ++kt;
      }
    }
    // segment 2: C2p channels from in2 (concat / broadcast lang)
    if (p.C2p) {
      const _Float16* base =
          p.bcast ? (p.in2 + (size_t)b * p.C2p + aoff)
                  : (ok ? (p.in2 + vox * p.C2p + aoff) : (p.zero + aoff));
      for (int c = 0; c < p.C2p; c += 32) {
        AFrag a;
        a.h[0] = *(const v8h*)(base + c);
        a.h[1] = *(const v8h*)(base + c + 16);
#pragma unroll
        for (int j = 0; j < NB; ++j) {
          v16h bf = *(const v16h*)(wpj[j] + (size_t)kt * 512);
          acc[j] = __builtin_amdgcn_wmma_f32_16x16x32_f16(
              false, a.v, false, bf, (short)0, acc[j], false, false);
        }
        ++kt;
      }
    }
  }

  // D layout: VGPR r -> M = grp*8 + r, N = lane & 15.
#pragma unroll
  for (int j = 0; j < NB; ++j) {
#pragma unroll
    for (int r = 0; r < 8; ++r) {
      float v = acc[j][r];
      if (p.relu) v = fmaxf(v, 0.0f);
      const size_t mo = (size_t)(mBase + (grp << 3) + r);
      p.out[mo * p.Cout + (size_t)(ntBase + j) * 16 + ml] = (_Float16)v;
    }
  }
}

// ---------------------------------------------------------------------------
// Pack fp32 weights [taps*(C1+C2), Cout] into padded WMMA B-fragment layout
// [NT][KT][32][16].  B layout (32x16 f16): lane L holds N=L&15, K=(L>>4)*8+h (h<8), 16+(L>>4)*8+(h-8) (h>=8).
// Padded channels (C1..C1p, C2..C2p) get zero weights.
// ---------------------------------------------------------------------------
__global__ void pack_w(const float* __restrict__ w, _Float16* __restrict__ wp,
                       int taps, int C1, int C2, int C1p, int C2p, int Cout,
                       int KT, size_t total) {
  size_t idx = (size_t)blockIdx.x * blockDim.x + threadIdx.x;
  if (idx >= total) return;
  const int h    = (int)(idx & 15);
  const int lane = (int)((idx >> 4) & 31);
  const size_t tile = idx >> 9;
  const int kt = (int)(tile % KT);
  const int nt = (int)(tile / KT);
  const int k  = kt * 32 + ((lane >> 4) << 3) + h + ((h >> 3) << 3);
  const int n  = nt * 16 + (lane & 15);
  const int Cinp = C1p + C2p;
  const int tap = k / Cinp;
  const int cp  = k - tap * Cinp;
  bool valid; int corig;
  if (cp < C1p) { valid = (cp < C1); corig = cp; }
  else { const int c2 = cp - C1p; valid = (c2 < C2); corig = C1 + c2; }
  float v = valid ? w[((size_t)tap * (C1 + C2) + corig) * Cout + n] : 0.0f;
  wp[idx] = (_Float16)v;
}

// Pad raw input (3 ch) to 32-channel fp16 buffer (zeros beyond ch 2).
__global__ void pad_input_f16(const float* __restrict__ x, _Float16* __restrict__ y,
                              size_t nvox) {
  size_t idx = (size_t)blockIdx.x * blockDim.x + threadIdx.x;
  if (idx >= nvox * 32) return;
  const int c = (int)(idx & 31);
  const size_t v = idx >> 5;
  y[idx] = (c < 3) ? (_Float16)x[v * 3 + c] : (_Float16)0.0f;
}

__global__ void f32_to_f16_k(const float* __restrict__ x, _Float16* __restrict__ y,
                             size_t n) {
  size_t i = (size_t)blockIdx.x * blockDim.x + threadIdx.x;
  if (i < n) y[i] = (_Float16)x[i];
}

__global__ void fill0_f16(_Float16* __restrict__ y, size_t n) {
  size_t i = (size_t)blockIdx.x * blockDim.x + threadIdx.x;
  if (i < n) y[i] = (_Float16)0.0f;
}

// 2x2x2 max pool; all dims and C are powers of two (shift/mask indexing).
// out dims D,H,W with logs lD,lH,lW; channels C with log lC.
__global__ void maxpool2_f16(const _Float16* __restrict__ in, _Float16* __restrict__ out,
                             int lD, int lH, int lW, int lC, size_t total) {
  size_t idx = (size_t)blockIdx.x * blockDim.x + threadIdx.x;
  if (idx >= total) return;
  const int C = 1 << lC, W = 1 << lW, H = 1 << lH, D = 1 << lD;
  const int c = (int)(idx & (C - 1));
  const int w = (int)((idx >> lC) & (W - 1));
  const int h = (int)((idx >> (lC + lW)) & (H - 1));
  const int d = (int)((idx >> (lC + lW + lH)) & (D - 1));
  const int b = (int)(idx >> (lC + lW + lH + lD));
  const int iH = 2 * H, iW = 2 * W;
  const size_t base = ((((size_t)b * 2 * D + 2 * d) * iH + 2 * h) * iW + 2 * w) * C + c;
  const size_t sW = (size_t)C, sH = (size_t)iW * C, sD = (size_t)iH * iW * C;
  float m = -3.0e38f;
#pragma unroll
  for (int dz = 0; dz < 2; ++dz)
#pragma unroll
    for (int dy = 0; dy < 2; ++dy)
#pragma unroll
      for (int dx = 0; dx < 2; ++dx) {
        float x = (float)in[base + dz * sD + dy * sH + dx * sW];
        m = x > m ? x : m;
      }
  out[idx] = (_Float16)m;
}

// Final 1x1x1 conv, 32 -> 1 channel, fp32 output, no relu.
__global__ void final_conv1(const _Float16* __restrict__ in, const float* __restrict__ w,
                            float* __restrict__ out, size_t M, int C) {
  size_t mI = (size_t)blockIdx.x * blockDim.x + threadIdx.x;
  if (mI >= M) return;
  float s = 0.0f;
  for (int c = 0; c < C; ++c) s += (float)in[mI * C + c] * w[c];
  out[mI] = s;
}

// ---------------------------------------------------------------------------
extern "C" void kernel_launch(void* const* d_in, const int* in_sizes, int n_in,
                              void* d_out, int out_size, void* d_ws, size_t ws_size,
                              hipStream_t stream) {
  (void)in_sizes; (void)n_in; (void)out_size; (void)ws_size;

  const float* input = (const float*)d_in[0];
  const float* lang  = (const float*)d_in[1];
  const float* Wf[12];
  for (int i = 0; i < 12; ++i) Wf[i] = (const float*)d_in[2 + i];
  const float* wfin = (const float*)d_in[14];
  float* out = (float*)d_out;

  char* ws = (char*)d_ws;
  size_t off = 0;
  auto alloc = [&](size_t bytes) -> char* {
    char* p = ws + off;
    off = (off + bytes + 1023) & ~(size_t)1023;
    return p;
  };

  // ---- per-layer configs: {taps, C1, C2, C1p, C2p, Cout} ------------------
  struct WCfg { int taps, C1, C2, C1p, C2p, Cout; };
  const WCfg wcfg[12] = {
    {27, 3,   0,   32,  0,   64},   // en1
    {27, 64,  0,   64,  0,   128},  // en2
    {27, 128, 0,   128, 0,   256},  // en3
    {1,  256, 256, 256, 256, 512},  // f1 (x3 ++ lang)
    {1,  512, 0,   512, 0,   384},  // f2
    {1,  384, 0,   384, 0,   256},  // f3
    {27, 256, 0,   256, 0,   256},  // d3t (deconv)
    {27, 256, 128, 256, 128, 256},  // d3c (up ++ x2)
    {27, 256, 0,   256, 0,   256},  // d2t
    {27, 256, 64,  256, 64,  128},  // d2c (up ++ x1)
    {27, 128, 0,   128, 0,   128},  // d1t
    {27, 128, 3,   128, 32,  32},   // d1c (up ++ padded input)
  };

  // ---- pack weights -------------------------------------------------------
  _Float16* wp[12];
  int layerKT[12];
  for (int i = 0; i < 12; ++i) {
    const int Cinp = wcfg[i].C1p + wcfg[i].C2p;
    const int KT = wcfg[i].taps * Cinp / 32;
    const int NT = wcfg[i].Cout / 16;
    layerKT[i] = KT;
    const size_t halfs = (size_t)NT * KT * 512;
    wp[i] = (_Float16*)alloc(halfs * 2);
    pack_w<<<dim3((unsigned)((halfs + 255) / 256)), 256, 0, stream>>>(
        Wf[i], wp[i], wcfg[i].taps, wcfg[i].C1, wcfg[i].C2,
        wcfg[i].C1p, wcfg[i].C2p, wcfg[i].Cout, KT, halfs);
  }

  // ---- activation buffers (fp16) -----------------------------------------
  _Float16* ZPG  = (_Float16*)alloc(1024 * 2);                     // zero page
  _Float16* IN16 = (_Float16*)alloc((size_t)2 * 262144 * 32 * 2);  // input, 32ch
  _Float16* LANG = (_Float16*)alloc((size_t)2 * 256 * 2);
  _Float16* X1   = (_Float16*)alloc((size_t)2 * 32768 * 64 * 2);   // skip 32^3
  _Float16* X2   = (_Float16*)alloc((size_t)2 * 4096 * 128 * 2);   // skip 16^3
  _Float16* X3   = (_Float16*)alloc((size_t)2 * 512 * 256 * 2);    // 8^3 (reused)
  _Float16* P    = (_Float16*)alloc((size_t)2 * 262144 * 128 * 2); // big ping
  _Float16* Q    = (_Float16*)alloc((size_t)2 * 262144 * 32 * 2);  // pong
  _Float16* R    = (_Float16*)alloc((size_t)2 * 512 * 384 * 2);    // fuse tmp

  fill0_f16<<<4, 256, 0, stream>>>(ZPG, 1024);
  pad_input_f16<<<(unsigned)(((size_t)524288 * 32 + 255) / 256), 256, 0, stream>>>(
      input, IN16, 524288);
  f32_to_f16_k<<<2, 256, 0, stream>>>(lang, LANG, 512);

  auto conv = [&](int li, const _Float16* in1, const _Float16* in2, _Float16* o,
                  int inD, int inH, int inW, int outD, int outH, int outW,
                  int mode, int bcast) {
    const WCfg& c = wcfg[li];
    ConvParams p;
    p.in1 = in1; p.in2 = in2; p.wp = wp[li]; p.zero = ZPG; p.out = o;
    p.inD = inD; p.inH = inH; p.inW = inW;
    p.outD = outD; p.outH = outH; p.outW = outW;
    p.lw = 31 - __builtin_clz((unsigned)outW);
    p.lh = 31 - __builtin_clz((unsigned)outH);
    p.ld = 31 - __builtin_clz((unsigned)outD);
    p.C1p = c.C1p; p.C2p = c.C2p; p.Cinp = c.C1p + c.C2p; p.Cout = c.Cout;
    p.taps = c.taps; p.mode = mode; p.bcast = bcast; p.relu = 1;
    p.KT = layerKT[li];
    const int M = 2 * outD * outH * outW;   // every layer: M % 128 == 0
    const int NB = (c.Cout % 64 == 0) ? 4 : 2;
    dim3 grid((unsigned)(M / 128), (unsigned)(c.Cout / (16 * NB)));
    if (NB == 4) wmma_conv<4><<<grid, 256, 0, stream>>>(p);
    else         wmma_conv<2><<<grid, 256, 0, stream>>>(p);
  };
  auto pool = [&](const _Float16* in, _Float16* o, int lS, int lC) {
    // pooled output dims are (1<<lS)^3, channels 1<<lC
    size_t total = (size_t)2 << (3 * lS + lC);
    maxpool2_f16<<<(unsigned)((total + 255) / 256), 256, 0, stream>>>(
        in, o, lS, lS, lS, lC, total);
  };

  // encoder
  conv(0, IN16, nullptr, P, 64, 64, 64, 64, 64, 64, 0, 0);
  pool(P, X1, 5, 6);                                        // -> 32^3 x 64
  conv(1, X1, nullptr, P, 32, 32, 32, 32, 32, 32, 0, 0);
  pool(P, X2, 4, 7);                                        // -> 16^3 x 128
  conv(2, X2, nullptr, P, 16, 16, 16, 16, 16, 16, 0, 0);
  pool(P, X3, 3, 8);                                        // -> 8^3 x 256
  // language fusion (1x1x1 convs; lang broadcast as concat channels)
  conv(3, X3, LANG, Q, 8, 8, 8, 8, 8, 8, 1, 1);
  conv(4, Q, nullptr, R, 8, 8, 8, 8, 8, 8, 1, 0);
  conv(5, R, nullptr, X3, 8, 8, 8, 8, 8, 8, 1, 0);
  // decoder: deconv(stride2)+relu, then conv over [up ++ skip]
  conv(6, X3, nullptr, P, 8, 8, 8, 16, 16, 16, 2, 0);      // d3t
  conv(7, P, X2, Q, 16, 16, 16, 16, 16, 16, 0, 0);         // d3c
  conv(8, Q, nullptr, P, 16, 16, 16, 32, 32, 32, 2, 0);    // d2t
  conv(9, P, X1, Q, 32, 32, 32, 32, 32, 32, 0, 0);         // d2c
  conv(10, Q, nullptr, P, 32, 32, 32, 64, 64, 64, 2, 0);   // d1t
  conv(11, P, IN16, Q, 64, 64, 64, 64, 64, 64, 0, 0);      // d1c
  // final 1x1x1 conv 32 -> 1, fp32 out, no relu
  final_conv1<<<(unsigned)((524288 + 255) / 256), 256, 0, stream>>>(
      Q, wfin, out, (size_t)524288, 32);
}
